// NonLocalBlock_84696755077241
// MI455X (gfx1250) — compile-verified, matrix operand order checked by reading, not run on hardware
//
#include <hip/hip_runtime.h>
#include <hip/hip_bf16.h>
#include <stdint.h>

// ---------------------------------------------------------------------------
// NonLocalBlock fused pipeline for MI455X (gfx1250, wave32, WMMA bf16, TDM)
//   B=16, C=256, Ci=128, H=W=64, N=4096, M=1024 (after 2x2 maxpool)
// K0a: pack theta_w/phi_w/g_w/W_w -> bf16
// K0b: x[B,C,N] f32 -> xT[B,N,C] bf16 (LDS tile transpose)
// K1 : xT -> theta[N,Ci], phi[M,Ci], g[Ci,M] bf16 (WMMA, LDS maxpool)
// K2 : flash attention y = softmax(theta phi^T) g^T (WMMA; phi/g tiles staged
//      into LDS by the Tensor Data Mover, double buffered)
// K3 : out = BN(W y + b) + x (WMMA)
// ---------------------------------------------------------------------------

#define B_  16
#define C_  256
#define CI_ 128
#define N_  4096
#define M_  1024

#ifndef __has_builtin
#define __has_builtin(x) 0
#endif

#if __has_builtin(__builtin_amdgcn_tensor_load_to_lds)
#define HAVE_TDM 1
#else
#define HAVE_TDM 0
#endif
#if defined(__clang_major__) && (__clang_major__ >= 23)
#define TDM_SIX 1
#else
#define TDM_SIX 0
#endif

typedef __attribute__((ext_vector_type(16))) __bf16 v16bf;
typedef __attribute__((ext_vector_type(8)))  float  v8f;
typedef unsigned int u32x4 __attribute__((ext_vector_type(4)));
typedef int          i32x4 __attribute__((ext_vector_type(4)));
typedef int          i32x8 __attribute__((ext_vector_type(8)));

union VA {              // one 16x32 bf16 A/B fragment per lane (8 VGPRs)
    v16bf    v;
    uint32_t u[8];
    uint4    q[2];
};

__device__ __forceinline__ v8f wmma_bf16(v16bf a, v16bf b, v8f c) {
    return __builtin_amdgcn_wmma_f32_16x16x32_bf16(
        false, a, false, b, (short)0, c, false, false);
}

__device__ __forceinline__ uint32_t pkbf(float a, float b) {
    union { __bf16 h[2]; uint32_t u; } t;
    t.h[0] = (__bf16)a; t.h[1] = (__bf16)b;
    return t.u;
}

__device__ __forceinline__ float rmax16(float v) {
    v = fmaxf(v, __shfl_xor(v, 1));
    v = fmaxf(v, __shfl_xor(v, 2));
    v = fmaxf(v, __shfl_xor(v, 4));
    v = fmaxf(v, __shfl_xor(v, 8));
    return v;
}
__device__ __forceinline__ float rsum16(float v) {
    v += __shfl_xor(v, 1);
    v += __shfl_xor(v, 2);
    v += __shfl_xor(v, 4);
    v += __shfl_xor(v, 8);
    return v;
}

// ---------------------------------------------------------------------------
// Tensor Data Mover helpers (CDNA5 D# per CDNA5 ISA)
// ---------------------------------------------------------------------------
#if HAVE_TDM
__device__ __forceinline__ void tdm_load_2d(uint32_t lds_addr, const void* gptr,
                                            uint32_t tile_d0, uint32_t tile_d1,
                                            uint32_t tensor_d0, uint32_t tensor_d1,
                                            uint32_t stride0_elems)
{
    const uint64_t ga = (uint64_t)(uintptr_t)gptr;
    u32x4 g0;
    g0[0] = 1u;                                           // count=1, no gather
    g0[1] = lds_addr;                                     // D#.lds_addr
    g0[2] = (uint32_t)ga;                                 // global_addr[31:0]
    g0[3] = (uint32_t)((ga >> 32) & 0x01FFFFFFu) | 0x80000000u; // [56:32] | type=2
    i32x8 g1;
    g1[0] = (int)(1u << 16);                              // data_size=1 -> 2 bytes
    g1[1] = (int)((tensor_d0 & 0xFFFFu) << 16);           // tensor_dim0[15:0]
    g1[2] = (int)((tensor_d0 >> 16) | ((tensor_d1 & 0xFFFFu) << 16));
    g1[3] = (int)((tensor_d1 >> 16) | (tile_d0 << 16));   // tile_dim0
    g1[4] = (int)(tile_d1 & 0xFFFFu);                     // tile_dim1 (tile_dim2=0)
    g1[5] = (int)stride0_elems;                           // tensor_dim0_stride lo
    g1[6] = 0;
    g1[7] = 0;
    const i32x4 z4 = { 0, 0, 0, 0 };
#if TDM_SIX
    const i32x8 z8 = { 0, 0, 0, 0, 0, 0, 0, 0 };
    __builtin_amdgcn_tensor_load_to_lds(g0, g1, z4, z4, z8, 0);
#else
    __builtin_amdgcn_tensor_load_to_lds(g0, g1, z4, z4, 0);
#endif
}
#endif

__device__ __forceinline__ void wait_tensor0(int tid) {
#if HAVE_TDM
    if (tid < 32) {
#if __has_builtin(__builtin_amdgcn_s_wait_tensorcnt)
        __builtin_amdgcn_s_wait_tensorcnt(0);
#else
        asm volatile("s_wait_tensorcnt 0x0" ::: "memory");
#endif
    }
#else
    (void)tid;
#endif
}

// Stage the shared phi tile (32 rows x Ci, contiguous) and g tile
// (Ci rows x 32 cols, row stride M) for one 32-wide M chunk into LDS.
__device__ __forceinline__ void stage_tiles(const __bf16* __restrict__ phi,
                                            const __bf16* __restrict__ g,
                                            __bf16* phd, __bf16* gld,
                                            int b, int jb, int tid)
{
#if HAVE_TDM
    if (tid < 32) {   // one TDM issue per block (wave 0)
        tdm_load_2d((uint32_t)(uintptr_t)phd,
                    phi + (size_t)(b * M_ + jb) * CI_,
                    4096u, 1u, 1u << 20, 1u, 4096u);              // 1-D, 8KB
        tdm_load_2d((uint32_t)(uintptr_t)gld,
                    g + (size_t)b * CI_ * M_ + jb,
                    32u, 128u, (uint32_t)M_, (uint32_t)CI_, (uint32_t)M_);
    }
#else
#pragma unroll
    for (int i = 0; i < 2; ++i) {
        const int idx = tid + i * 256;                    // 512 uint4 per tile
        ((uint4*)phd)[idx] =
            ((const uint4*)(phi + (size_t)(b * M_ + jb) * CI_))[idx];
        const int row = idx >> 2, q = idx & 3;
        ((uint4*)(gld + row * 32))[q] =
            ((const uint4*)(g + ((size_t)(b * CI_ + row)) * M_ + jb))[q];
    }
#endif
}

// ---------------------------------------------------------------------------
// K0a: pack the four weight matrices to bf16 (wpack: 4 x 32768 elems)
// ---------------------------------------------------------------------------
__global__ __launch_bounds__(256) void k0_packw(
    const float* __restrict__ tw, const float* __restrict__ pw,
    const float* __restrict__ gw, const float* __restrict__ Ww,
    __bf16* __restrict__ wpack)
{
    const int id  = blockIdx.x * 256 + threadIdx.x;   // 65536 dwords total
    const int arr = id >> 14;
    const int off = (id & 16383) * 2;
    const float* s = (arr == 0) ? tw : (arr == 1) ? pw : (arr == 2) ? gw : Ww;
    *(uint32_t*)(wpack + (size_t)arr * 32768 + off) = pkbf(s[off], s[off + 1]);
}

// ---------------------------------------------------------------------------
// K0b: transpose x[B,C,N] f32 -> xT[B,N,C] bf16 through LDS.
// grid = B * (C/32) * (N/64) = 8192 blocks.
// ---------------------------------------------------------------------------
__global__ __launch_bounds__(256) void k0_transpose(
    const float* __restrict__ x, __bf16* __restrict__ xT)
{
    __shared__ float t[32][65];   // +1 pad: conflict-free column reads
    const int tid = threadIdx.x;
    const int blk = blockIdx.x;
    const int b   = blk >> 9;
    const int r   = blk & 511;
    const int c0  = (r >> 6) * 32;
    const int n0  = (r & 63) * 64;

    const float* xb = x + ((size_t)b * C_ + c0) * N_ + n0;
    const int cl = tid >> 6;      // 0..3
    const int nl = tid & 63;
#pragma unroll
    for (int i = 0; i < 8; ++i)
        t[cl + 4 * i][nl] = xb[(size_t)(cl + 4 * i) * N_ + nl];
    __syncthreads();

#pragma unroll
    for (int i = 0; i < 4; ++i) {
        const int flat = tid + i * 256;   // 64 n * 16 dwords
        const int n  = flat >> 4;
        const int dw = flat & 15;
        const int c  = dw * 2;
        *(uint32_t*)(xT + ((size_t)(b * N_ + n0 + n) * C_ + c0 + c)) =
            pkbf(t[c][n], t[c + 1][n]);
    }
}

// ---------------------------------------------------------------------------
// K1: projections. grid = B*(H/2) = 512 blocks, 8 waves. Block owns one
// row pair (128 pixels); wave owns a 16-pixel A tile (all K=256 in regs).
// ---------------------------------------------------------------------------
__global__ __launch_bounds__(256) void k1_proj(
    const __bf16* __restrict__ xT, const __bf16* __restrict__ wpack,
    const float* __restrict__ tb, const float* __restrict__ pb,
    const float* __restrict__ gb,
    __bf16* __restrict__ theta, __bf16* __restrict__ phi,
    __bf16* __restrict__ gbuf)
{
    __shared__ __bf16 phl[128 * 128];
    __shared__ __bf16 ghl[128 * 128];

    const int tid  = threadIdx.x;
    const int wave = tid >> 5;
    const int lane = tid & 31;
    const int ln   = lane & 15;
    const int hi   = lane >> 4;

    const int blk = blockIdx.x;
    const int b   = blk >> 5;
    const int h0  = blk & 31;
    const int n0  = h0 * 128;
    const int pt  = n0 + wave * 16;

    // A fragments straight from bf16 xT rows (two b128 per 32-K chunk)
    VA a[8];
    {
        const uint4* rowq = (const uint4*)(xT + (size_t)(b * N_ + pt + ln) * C_);
#pragma unroll
        for (int kc = 0; kc < 8; ++kc) {
            a[kc].q[0] = rowq[4 * kc + hi];
            a[kc].q[1] = rowq[4 * kc + hi + 2];
        }
    }

    const float* bsel[3] = { tb, pb, gb };
    const v8f Zv = { 0.f, 0.f, 0.f, 0.f, 0.f, 0.f, 0.f, 0.f };

    for (int p = 0; p < 3; ++p) {
        const __bf16* wp = wpack + (size_t)p * 32768;
        const float*  bp = bsel[p];
#pragma unroll 2
        for (int ot = 0; ot < 8; ++ot) {
            const int o = ot * 16 + ln;                 // output channel
            const uint4* wq = (const uint4*)(wp + (size_t)o * C_);
            if (ot + 1 < 8)
                __builtin_prefetch(wp + (size_t)(o + 16) * C_, 0, 1);
            v8f acc = Zv;
#pragma unroll
            for (int kc = 0; kc < 8; ++kc) {
                VA bf;
                bf.q[0] = wq[4 * kc + hi];
                bf.q[1] = wq[4 * kc + hi + 2];
                acc = wmma_bf16(a[kc].v, bf.v, acc);
            }
            const float bias = bp[o];
            if (p == 0) {
#pragma unroll
                for (int r = 0; r < 8; ++r) {
                    const int nn = pt + r + 8 * hi;
                    theta[(size_t)(b * N_ + nn) * CI_ + o] = (__bf16)(acc[r] + bias);
                }
            } else {
                __bf16* L = (p == 1) ? phl : ghl;
#pragma unroll
                for (int r = 0; r < 8; ++r) {
                    const int lp = wave * 16 + r + 8 * hi;
                    L[lp * 128 + o] = (__bf16)(acc[r] + bias);
                }
            }
        }
    }

    __syncthreads();

    // 2x2 max pool; pooled row width = 32
    for (int t = tid; t < 32 * 128; t += 256) {
        const int pwi = t >> 7;
        const int o   = t & 127;
        const int l00 = (2 * pwi) * 128 + o;
        const int l01 = (2 * pwi + 1) * 128 + o;
        const int l10 = (64 + 2 * pwi) * 128 + o;
        const int l11 = (65 + 2 * pwi) * 128 + o;
        const int pm  = h0 * 32 + pwi;
        const float mp = fmaxf(fmaxf((float)phl[l00], (float)phl[l01]),
                               fmaxf((float)phl[l10], (float)phl[l11]));
        const float mg = fmaxf(fmaxf((float)ghl[l00], (float)ghl[l01]),
                               fmaxf((float)ghl[l10], (float)ghl[l11]));
        phi [(size_t)(b * M_ + pm) * CI_ + o] = (__bf16)mp;   // [M, Ci]
        gbuf[(size_t)(b * CI_ + o) * M_ + pm] = (__bf16)mg;   // [Ci, M]
    }
}

// ---------------------------------------------------------------------------
// K2: fused flash attention. grid = B*(N/128) = 512 blocks, 8 waves.
// phi/g tiles DMA-staged into LDS (double buffered, TDM), P transposed
// through per-wave LDS bounce, 16 WMMAs per 32-wide M chunk.
// ---------------------------------------------------------------------------
__global__ __launch_bounds__(256) void k2_attn(
    const __bf16* __restrict__ theta, const __bf16* __restrict__ phi,
    const __bf16* __restrict__ g,     __bf16* __restrict__ ybuf)
{
    __shared__ __align__(16) __bf16 phl2[2][32 * CI_];   // 2 x 8 KB
    __shared__ __align__(16) __bf16 gl2 [2][CI_ * 32];   // 2 x 8 KB
    __shared__ __align__(16) __bf16 plds[8 * 512];       // P transpose, 8 KB

    const int tid  = threadIdx.x;
    const int wave = tid >> 5;
    const int lane = tid & 31;
    const int ln   = lane & 15;
    const int hi   = lane >> 4;

    const int blk   = blockIdx.x;
    const int b     = blk >> 5;
    const int ntile = (blk & 31) * 128 + wave * 16;

    // theta A fragments: 16 rows x K=128 resident in registers
    VA at[4];
    {
        const uint4* rowq = (const uint4*)(theta + (size_t)(b * N_ + ntile + ln) * CI_);
#pragma unroll
        for (int kc = 0; kc < 4; ++kc) {
            at[kc].q[0] = rowq[4 * kc + hi];
            at[kc].q[1] = rowq[4 * kc + hi + 2];
        }
    }

    const v8f Zv = { 0.f, 0.f, 0.f, 0.f, 0.f, 0.f, 0.f, 0.f };
    v8f yacc[8];
#pragma unroll
    for (int ct = 0; ct < 8; ++ct) yacc[ct] = Zv;
    float mrow[8], lrow[8];
#pragma unroll
    for (int r = 0; r < 8; ++r) { mrow[r] = -1e30f; lrow[r] = 0.f; }

    __bf16* pl = plds + wave * 512;

    stage_tiles(phi, g, phl2[0], gl2[0], b, 0, tid);     // prologue DMA

    for (int it = 0; it < M_ / 32; ++it) {
        const int cur = it & 1;
        wait_tensor0(tid);            // tile `it` landed in LDS (wave 0)
        __syncthreads();              // everyone sees it; other buffer is free
        if (it + 1 < M_ / 32)         // overlap DMA of tile it+1 with compute
            stage_tiles(phi, g, phl2[1 - cur], gl2[1 - cur], b, (it + 1) * 32, tid);

        // ---- S = theta * phi^T : 8 WMMAs from LDS fragments ----
        v8f s0 = Zv, s1 = Zv;
        {
            const uint4* rq0 = (const uint4*)(phl2[cur] + ln * CI_);
            const uint4* rq1 = (const uint4*)(phl2[cur] + (16 + ln) * CI_);
#pragma unroll
            for (int kc = 0; kc < 4; ++kc) {
                VA pb;
                pb.q[0] = rq0[4 * kc + hi];
                pb.q[1] = rq0[4 * kc + hi + 2];
                s0 = wmma_bf16(at[kc].v, pb.v, s0);
            }
#pragma unroll
            for (int kc = 0; kc < 4; ++kc) {
                VA pb;
                pb.q[0] = rq1[4 * kc + hi];
                pb.q[1] = rq1[4 * kc + hi + 2];
                s1 = wmma_bf16(at[kc].v, pb.v, s1);
            }
        }

        // ---- online softmax (16-lane butterflies, wave32-safe) ----
#pragma unroll
        for (int r = 0; r < 8; ++r) {
            const float v  = rmax16(fmaxf(s0[r], s1[r]));
            const float nm = fmaxf(mrow[r], v);
            const float alpha = __expf(mrow[r] - nm);
            const float p0 = __expf(s0[r] - nm);
            const float p1 = __expf(s1[r] - nm);
            s0[r] = p0; s1[r] = p1;
            lrow[r] = lrow[r] * alpha + rsum16(p0 + p1);
            mrow[r] = nm;
#pragma unroll
            for (int ct = 0; ct < 8; ++ct) yacc[ct][r] *= alpha;
        }

        // ---- transpose P (C/D layout -> A layout) through LDS ----
#pragma unroll
        for (int r = 0; r < 8; ++r) {
            const int row = r + 8 * hi;
            pl[row * 32 + ln]      = (__bf16)s0[r];
            pl[row * 32 + 16 + ln] = (__bf16)s1[r];
        }
        asm volatile("s_wait_dscnt 0x0" ::: "memory");
        VA pa;
        {
            const uint4* rq = (const uint4*)(pl + ln * 32);
            pa.q[0] = rq[hi];
            pa.q[1] = rq[hi + 2];
        }

        // ---- y += P * g^T : 8 WMMAs from the DMA-staged g tile ----
#pragma unroll
        for (int ct = 0; ct < 8; ++ct) {
            const uint4* rq = (const uint4*)(gl2[cur] + (ct * 16 + ln) * 32);
            VA gb;
            gb.q[0] = rq[hi];
            gb.q[1] = rq[hi + 2];
            yacc[ct] = wmma_bf16(pa.v, gb.v, yacc[ct]);
        }
    }

    // ---- finalize: divide by row sums, store y[N,Ci] bf16 ----
#pragma unroll
    for (int r = 0; r < 8; ++r) {
        const float inv = 1.0f / lrow[r];
        const int   nn  = ntile + r + 8 * hi;
        __bf16* yr = ybuf + (size_t)(b * N_ + nn) * CI_;
#pragma unroll
        for (int ct = 0; ct < 8; ++ct)
            yr[ct * 16 + ln] = (__bf16)(yacc[ct][r] * inv);
    }
}

// ---------------------------------------------------------------------------
// K3: out = BN(W y + b) + x. grid = B*64*2 = 2048 blocks, 8 waves.
// A = W rows (lane = out channel), B = y^T tiles; D cols = pixels ->
// coalesced fp32 residual load/store.
// ---------------------------------------------------------------------------
__global__ __launch_bounds__(256) void k3_out(
    const float* __restrict__ x,  const __bf16* __restrict__ wpack,
    const float* __restrict__ Wb, const float* __restrict__ gamma,
    const float* __restrict__ beta, const float* __restrict__ mean,
    const float* __restrict__ var, const __bf16* __restrict__ ybuf,
    float* __restrict__ out)
{
    const int tid  = threadIdx.x;
    const int wave = tid >> 5;
    const int lane = tid & 31;
    const int ln   = lane & 15;
    const int hi   = lane >> 4;

    const int idx    = blockIdx.x;
    const int b      = idx >> 7;
    const int rem    = idx & 127;
    const int nblock = (rem >> 1) * 64;
    const int c0     = (((rem & 1) * 8) + wave) * 16;

    // A fragments from packed bf16 W_w rows (K = Ci = 128)
    VA wa[4];
    {
        const uint4* wq = (const uint4*)(wpack + (size_t)3 * 32768 + (size_t)(c0 + ln) * CI_);
#pragma unroll
        for (int kc = 0; kc < 4; ++kc) {
            wa[kc].q[0] = wq[4 * kc + hi];
            wa[kc].q[1] = wq[4 * kc + hi + 2];
        }
    }

    float sc[8], sh[8], wb[8];
#pragma unroll
    for (int r = 0; r < 8; ++r) {
        const int cr = c0 + r + 8 * hi;
        const float s = gamma[cr] * rsqrtf(var[cr] + 1e-5f);
        sc[r] = s;
        sh[r] = beta[cr] - mean[cr] * s;
        wb[r] = Wb[cr];
    }

    const v8f Zv = { 0.f, 0.f, 0.f, 0.f, 0.f, 0.f, 0.f, 0.f };
#pragma unroll 1
    for (int nt = 0; nt < 4; ++nt) {
        const int n0 = nblock + nt * 16;
        const uint4* rq = (const uint4*)(ybuf + (size_t)(b * N_ + n0 + ln) * CI_);
        v8f acc = Zv;
#pragma unroll
        for (int kc = 0; kc < 4; ++kc) {
            VA yb;
            yb.q[0] = rq[4 * kc + hi];
            yb.q[1] = rq[4 * kc + hi + 2];
            acc = wmma_bf16(wa[kc].v, yb.v, acc);
        }
#pragma unroll
        for (int r = 0; r < 8; ++r) {
            const int cr = c0 + r + 8 * hi;
            const size_t xi = ((size_t)(b * C_ + cr)) * N_ + n0 + ln;
            out[xi] = (acc[r] + wb[r]) * sc[r] + sh[r] + x[xi];
        }
    }
}

// ---------------------------------------------------------------------------
extern "C" void kernel_launch(void* const* d_in, const int* in_sizes, int n_in,
                              void* d_out, int out_size, void* d_ws, size_t ws_size,
                              hipStream_t stream)
{
    (void)in_sizes; (void)n_in; (void)out_size; (void)ws_size;
    const float* x     = (const float*)d_in[0];
    const float* tw    = (const float*)d_in[1];
    const float* tb    = (const float*)d_in[2];
    const float* pw    = (const float*)d_in[3];
    const float* pb    = (const float*)d_in[4];
    const float* gw    = (const float*)d_in[5];
    const float* gb    = (const float*)d_in[6];
    const float* Ww    = (const float*)d_in[7];
    const float* Wb    = (const float*)d_in[8];
    const float* gamma = (const float*)d_in[9];
    const float* beta  = (const float*)d_in[10];
    const float* mean  = (const float*)d_in[11];
    const float* var   = (const float*)d_in[12];

    char* ws = (char*)d_ws;
    __bf16* xT    = (__bf16*)(ws);                  // 32 MiB [B,N,C]
    __bf16* theta = (__bf16*)(ws + (32u << 20));    // 16 MiB [B,N,Ci]
    __bf16* phi   = (__bf16*)(ws + (48u << 20));    //  4 MiB [B,M,Ci]
    __bf16* gbuf  = (__bf16*)(ws + (52u << 20));    //  4 MiB [B,Ci,M]
    __bf16* ybuf  = (__bf16*)(ws + (56u << 20));    // 16 MiB [B,N,Ci]
    __bf16* wpack = (__bf16*)(ws + (72u << 20));    // 256 KiB packed weights

    k0_packw<<<256, 256, 0, stream>>>(tw, pw, gw, Ww, wpack);
    k0_transpose<<<B_ * 512, 256, 0, stream>>>(x, xT);
    k1_proj<<<B_ * 32, 256, 0, stream>>>(xT, wpack, tb, pb, gb, theta, phi, gbuf);
    k2_attn<<<B_ * 32, 256, 0, stream>>>(theta, phi, gbuf, ybuf);
    k3_out<<<B_ * 128, 256, 0, stream>>>(x, wpack, Wb, gamma, beta, mean, var,
                                         ybuf, (float*)d_out);
}
